// OctreeGnResBlock2_24988119728285
// MI455X (gfx1250) — matrix operand order repaired and checked by reading, not compiled, not run on hardware
//
#include <hip/hip_runtime.h>
#include <hip/hip_bf16.h>

// CDNA5 / gfx1250: wave32, WMMA 16x16x32 bf16 with fp32 accumulate.
// bf16 hi/lo split-precision GEMM (3 WMMAs per product) for near-fp32 accuracy.
// Each wave owns TWO 16-node M-tiles so weight (B) fragments are reused,
// doubling WMMA density per load and widening same-acc dependency distance.

typedef __attribute__((ext_vector_type(16))) __bf16 v16bf;
typedef __attribute__((ext_vector_type(8)))  __bf16 v8bf;
typedef __attribute__((ext_vector_type(8)))  float  v8f;

#define KTAPS 27
#define CIN   32
#define COUT  64

__device__ __forceinline__ v8f wmma_bf16(v16bf a, v16bf b, v8f c) {
    return __builtin_amdgcn_wmma_f32_16x16x32_bf16(
        /*neg_a=*/false, a, /*neg_b=*/false, b,
        /*c_mod=*/(short)0, c, /*reuse_a=*/false, /*reuse_b=*/false);
}

// A-fragment for 16x16x32 bf16 WMMA from a row-major bf16 row.
// lanes 0-15: K = koff+0..7 (v0..3), K = 16+koff+0..7 (v4..7); koff = 8 for lanes>=16.
__device__ __forceinline__ v16bf load_a(const __bf16* __restrict__ row, int koff) {
    v8bf c0 = *(const v8bf*)(row + koff);
    v8bf c1 = *(const v8bf*)(row + 16 + koff);
    return __builtin_shufflevector(c0, c1,
        0, 1, 2, 3, 4, 5, 6, 7, 8, 9, 10, 11, 12, 13, 14, 15);
}

// 4-lane cross-lane ops via ds_swizzle group-of-32 immediates:
// SWAPX1 = and 0x1f | xor 1 -> 0x7C01 ; SWAPX2 -> 0x7C02. One DS op each.
__device__ __forceinline__ float swz_xor1(float x) {
    return __builtin_bit_cast(float,
        __builtin_amdgcn_ds_swizzle(__builtin_bit_cast(int, x), 0x7C01));
}
__device__ __forceinline__ float swz_xor2(float x) {
    return __builtin_bit_cast(float,
        __builtin_amdgcn_ds_swizzle(__builtin_bit_cast(int, x), 0x7C02));
}

// GroupNorm over a 4-channel group: channels of one group live in 4 adjacent
// lanes of the same half-wave -> 2x ds_swizzle reduction, no LDS traffic.
__device__ __forceinline__ float gn4(float x) {
    float s = x + swz_xor1(x);
    s += swz_xor2(s);
    float xx = x * x;
    float q = xx + swz_xor1(xx);
    q += swz_xor2(q);
    float mean = s * 0.25f;
    float var  = q * 0.25f - mean * mean;
    return (x - mean) * rsqrtf(var + 1e-5f);
}

// fp32 -> bf16 hi/lo split (a = hi + lo, product err ~2^-18)
__global__ void cvt_split_kernel(const float* __restrict__ x,
                                 __bf16* __restrict__ hi, __bf16* __restrict__ lo,
                                 long long count) {
    long long i = (long long)blockIdx.x * blockDim.x + threadIdx.x;
    long long stride = (long long)gridDim.x * blockDim.x;
    for (; i < count; i += stride) {
        float v = x[i];
        __bf16 h = (__bf16)v;
        hi[i] = h;
        lo[i] = (__bf16)(v - (float)h);
    }
}

// Pre-swizzle weights W[taps][kdim][64] into B-fragment layout:
// frag f = ((k*(kdim/32)+kk)*4 + t); lane l holds col n = 16t + (l&15),
// K = (l<16?0:8)+e for e<8, 16+(l<16?0:8)+(e-8) for e>=8 -> one aligned 32B chunk per lane.
__global__ void pack_b_kernel(const float* __restrict__ W,
                              __bf16* __restrict__ hi, __bf16* __restrict__ lo,
                              int taps, int kdim) {
    int idx = blockIdx.x * blockDim.x + threadIdx.x;
    int khalves = kdim >> 5;
    int total = taps * khalves * 4 * 512;
    if (idx >= total) return;
    int e    = idx & 15;
    int lane = (idx >> 4) & 31;
    int f    = idx >> 9;
    int t    = f & 3;
    int rest = f >> 2;
    int kk   = rest % khalves;
    int k    = rest / khalves;
    int n    = (t << 4) + (lane & 15);
    int kr   = kk * 32 + ((lane >> 4) << 3) + (e < 8 ? e : 8 + e);
    float v  = W[((long long)k * kdim + kr) * COUT + n];
    __bf16 h = (__bf16)v;
    hi[idx] = h;
    lo[idx] = (__bf16)(v - (float)h);
}

// conv3x3a: gathered WMMA GEMM -> GroupNorm -> ReLU -> h (bf16 hi/lo)
__global__ __launch_bounds__(256) void conv_a_kernel(
    const __bf16* __restrict__ x_hi, const __bf16* __restrict__ x_lo,
    const int* __restrict__ neigh,
    const __bf16* __restrict__ wa_hi, const __bf16* __restrict__ wa_lo,
    const float* __restrict__ gw_, const float* __restrict__ gb_,
    __bf16* __restrict__ h_hi, __bf16* __restrict__ h_lo, int n) {
    const int lane = threadIdx.x & 31;
    const int wid  = blockIdx.x * (blockDim.x >> 5) + (threadIdx.x >> 5);
    const int i0   = wid << 5;                 // 32 nodes / wave
    if (i0 >= n) return;
    const int m    = lane & 15;
    const int half = lane >> 4;
    const int koff = half << 3;

    v8f acc[2][4] = {};
    const int* nrow0 = neigh + (long long)(i0 + m) * KTAPS;
    const int* nrow1 = neigh + (long long)min(i0 + 16 + m, n - 1) * KTAPS;

    for (int k = 0; k < KTAPS; ++k) {
        const long long g0 = nrow0[k];
        const long long g1 = nrow1[k];
        v16bf a0h = load_a(x_hi + g0 * CIN, koff);
        v16bf a0l = load_a(x_lo + g0 * CIN, koff);
        v16bf a1h = load_a(x_hi + g1 * CIN, koff);
        v16bf a1l = load_a(x_lo + g1 * CIN, koff);
        const __bf16* bhp = wa_hi + (((k * 4) * 32 + lane) << 4);
        const __bf16* blp = wa_lo + (((k * 4) * 32 + lane) << 4);
        v16bf bh[4], bl[4];
#pragma unroll
        for (int t = 0; t < 4; ++t) {
            bh[t] = *(const v16bf*)(bhp + t * 512);
            bl[t] = *(const v16bf*)(blp + t * 512);
        }
        // same-acc WMMA deps are 8 ops apart
#pragma unroll
        for (int t = 0; t < 4; ++t) acc[0][t] = wmma_bf16(a0h, bh[t], acc[0][t]);
#pragma unroll
        for (int t = 0; t < 4; ++t) acc[1][t] = wmma_bf16(a1h, bh[t], acc[1][t]);
#pragma unroll
        for (int t = 0; t < 4; ++t) acc[0][t] = wmma_bf16(a0l, bh[t], acc[0][t]);
#pragma unroll
        for (int t = 0; t < 4; ++t) acc[1][t] = wmma_bf16(a1l, bh[t], acc[1][t]);
#pragma unroll
        for (int t = 0; t < 4; ++t) acc[0][t] = wmma_bf16(a0h, bl[t], acc[0][t]);
#pragma unroll
        for (int t = 0; t < 4; ++t) acc[1][t] = wmma_bf16(a1h, bl[t], acc[1][t]);
    }

    // GroupNorm (16 groups of 4 ch) + ReLU, emit bf16 hi/lo of h
#pragma unroll
    for (int tile = 0; tile < 2; ++tile) {
        const int base = i0 + tile * 16;
        const bool live = base < n;
#pragma unroll
        for (int t = 0; t < 4; ++t) {
            const int c = (t << 4) + m;
            const float w = gw_[c], b = gb_[c];
#pragma unroll
            for (int r = 0; r < 8; ++r) {
                float y = fmaxf(gn4(acc[tile][t][r]) * w + b, 0.0f);
                if (live) {
                    long long idx = (long long)(base + (half << 3) + r) * COUT + c;
                    __bf16 yh = (__bf16)y;
                    h_hi[idx] = yh;
                    h_lo[idx] = (__bf16)(y - (float)yh);
                }
            }
        }
    }
}

// conv3x3b + 1x1 shortcut: fused WMMA GEMMs -> two GroupNorms -> add -> ReLU
__global__ __launch_bounds__(256) void conv_b_kernel(
    const __bf16* __restrict__ h_hi, const __bf16* __restrict__ h_lo,
    const __bf16* __restrict__ x_hi, const __bf16* __restrict__ x_lo,
    const int* __restrict__ neigh,
    const __bf16* __restrict__ wb_hi, const __bf16* __restrict__ wb_lo,
    const __bf16* __restrict__ w1_hi, const __bf16* __restrict__ w1_lo,
    const float* __restrict__ gbw, const float* __restrict__ gbb,
    const float* __restrict__ gsw, const float* __restrict__ gsb,
    float* __restrict__ out, int n) {
    const int lane = threadIdx.x & 31;
    const int wid  = blockIdx.x * (blockDim.x >> 5) + (threadIdx.x >> 5);
    const int i0   = wid << 5;                 // 32 nodes / wave
    if (i0 >= n) return;
    const int m    = lane & 15;
    const int half = lane >> 4;
    const int koff = half << 3;

    v8f acc[2][4] = {};
    const int* nrow0 = neigh + (long long)(i0 + m) * KTAPS;
    const int* nrow1 = neigh + (long long)min(i0 + 16 + m, n - 1) * KTAPS;

    for (int k = 0; k < KTAPS; ++k) {
        const long long g0 = nrow0[k];
        const long long g1 = nrow1[k];
#pragma unroll
        for (int kk = 0; kk < 2; ++kk) {       // K = 64 -> two K=32 steps
            v16bf a0h = load_a(h_hi + g0 * COUT + kk * 32, koff);
            v16bf a0l = load_a(h_lo + g0 * COUT + kk * 32, koff);
            v16bf a1h = load_a(h_hi + g1 * COUT + kk * 32, koff);
            v16bf a1l = load_a(h_lo + g1 * COUT + kk * 32, koff);
            const long long fid = ((long long)k * 2 + kk) * 4;
            const __bf16* bhp = wb_hi + ((fid * 32 + lane) << 4);
            const __bf16* blp = wb_lo + ((fid * 32 + lane) << 4);
            v16bf bh[4], bl[4];
#pragma unroll
            for (int t = 0; t < 4; ++t) {
                bh[t] = *(const v16bf*)(bhp + t * 512);
                bl[t] = *(const v16bf*)(blp + t * 512);
            }
#pragma unroll
            for (int t = 0; t < 4; ++t) acc[0][t] = wmma_bf16(a0h, bh[t], acc[0][t]);
#pragma unroll
            for (int t = 0; t < 4; ++t) acc[1][t] = wmma_bf16(a1h, bh[t], acc[1][t]);
#pragma unroll
            for (int t = 0; t < 4; ++t) acc[0][t] = wmma_bf16(a0l, bh[t], acc[0][t]);
#pragma unroll
            for (int t = 0; t < 4; ++t) acc[1][t] = wmma_bf16(a1l, bh[t], acc[1][t]);
#pragma unroll
            for (int t = 0; t < 4; ++t) acc[0][t] = wmma_bf16(a0h, bl[t], acc[0][t]);
#pragma unroll
            for (int t = 0; t < 4; ++t) acc[1][t] = wmma_bf16(a1h, bl[t], acc[1][t]);
        }
    }

    // shortcut GEMM after the main loop so its accumulators don't overlap
    // the B-fragment registers: sc = x[own rows] @ W1, K = 32
    v8f scc[2][4] = {};
#pragma unroll
    for (int tile = 0; tile < 2; ++tile) {
        const long long rr = min(i0 + tile * 16 + m, n - 1);
        v16bf ah = load_a(x_hi + rr * CIN, koff);
        v16bf al = load_a(x_lo + rr * CIN, koff);
#pragma unroll
        for (int t = 0; t < 4; ++t) {
            v16bf bh = *(const v16bf*)(w1_hi + ((t * 32 + lane) << 4));
            v16bf bl = *(const v16bf*)(w1_lo + ((t * 32 + lane) << 4));
            scc[tile][t] = wmma_bf16(ah, bh, scc[tile][t]);
            scc[tile][t] = wmma_bf16(al, bh, scc[tile][t]);
            scc[tile][t] = wmma_bf16(ah, bl, scc[tile][t]);
        }
    }

    // out = relu(GN_b(conv) + GN_s(shortcut)), fp32
#pragma unroll
    for (int tile = 0; tile < 2; ++tile) {
        const int base = i0 + tile * 16;
        const bool live = base < n;
#pragma unroll
        for (int t = 0; t < 4; ++t) {
            const int c = (t << 4) + m;
            const float wbv = gbw[c], bbv = gbb[c];
            const float wsv = gsw[c], bsv = gsb[c];
#pragma unroll
            for (int r = 0; r < 8; ++r) {
                float y1 = gn4(acc[tile][t][r]) * wbv + bbv;
                float y2 = gn4(scc[tile][t][r]) * wsv + bsv;
                if (live)
                    out[(long long)(base + (half << 3) + r) * COUT + c] =
                        fmaxf(y1 + y2, 0.0f);
            }
        }
    }
}

extern "C" void kernel_launch(void* const* d_in, const int* in_sizes, int n_in,
                              void* d_out, int out_size, void* d_ws, size_t ws_size,
                              hipStream_t stream) {
    (void)n_in; (void)out_size; (void)ws_size;
    const float* data = (const float*)d_in[0];
    const int*   neigh = (const int*)d_in[1];
    const float* Wa   = (const float*)d_in[2];
    const float* ga_w = (const float*)d_in[3];
    const float* ga_b = (const float*)d_in[4];
    const float* Wb   = (const float*)d_in[5];
    const float* gb_w = (const float*)d_in[6];
    const float* gb_b = (const float*)d_in[7];
    const float* W1   = (const float*)d_in[8];
    const float* gs_w = (const float*)d_in[9];
    const float* gs_b = (const float*)d_in[10];
    float* out = (float*)d_out;
    const int N = in_sizes[0] / CIN;

    // workspace carve-out (256B aligned)
    char* ws = (char*)d_ws;
    size_t off = 0;
    auto carve = [&](size_t bytes) -> char* {
        char* p = ws + off;
        off += (bytes + 255) & ~(size_t)255;
        return p;
    };
    __bf16* x_hi = (__bf16*)carve((size_t)N * CIN * 2);
    __bf16* x_lo = (__bf16*)carve((size_t)N * CIN * 2);
    __bf16* h_hi = (__bf16*)carve((size_t)N * COUT * 2);
    __bf16* h_lo = (__bf16*)carve((size_t)N * COUT * 2);
    const int wa_elems = KTAPS * 1 * 4 * 512;
    const int wb_elems = KTAPS * 2 * 4 * 512;
    const int w1_elems = 1 * 1 * 4 * 512;
    __bf16* wa_hi = (__bf16*)carve((size_t)wa_elems * 2);
    __bf16* wa_lo = (__bf16*)carve((size_t)wa_elems * 2);
    __bf16* wb_hi = (__bf16*)carve((size_t)wb_elems * 2);
    __bf16* wb_lo = (__bf16*)carve((size_t)wb_elems * 2);
    __bf16* w1_hi = (__bf16*)carve((size_t)w1_elems * 2);
    __bf16* w1_lo = (__bf16*)carve((size_t)w1_elems * 2);

    // 1) split input features to bf16 hi/lo
    {
        long long cnt = (long long)N * CIN;
        int blocks = (int)((cnt + 255) / 256);
        cvt_split_kernel<<<blocks, 256, 0, stream>>>(data, x_hi, x_lo, cnt);
    }
    // 2) pre-swizzle weights into WMMA B-fragment layout
    pack_b_kernel<<<(wa_elems + 255) / 256, 256, 0, stream>>>(Wa, wa_hi, wa_lo, KTAPS, CIN);
    pack_b_kernel<<<(wb_elems + 255) / 256, 256, 0, stream>>>(Wb, wb_hi, wb_lo, KTAPS, COUT);
    pack_b_kernel<<<(w1_elems + 255) / 256, 256, 0, stream>>>(W1, w1_hi, w1_lo, 1, CIN);

    // 3) conv3x3a -> GN -> ReLU -> h (bf16 hi/lo)
    const int waves  = (N + 31) / 32;
    const int blocks = (waves * 32 + 255) / 256;
    conv_a_kernel<<<blocks, 256, 0, stream>>>(x_hi, x_lo, neigh, wa_hi, wa_lo,
                                              ga_w, ga_b, h_hi, h_lo, N);
    // 4) conv3x3b -> GN, shortcut 1x1 -> GN, add, ReLU -> out (fp32)
    conv_b_kernel<<<blocks, 256, 0, stream>>>(h_hi, h_lo, x_hi, x_lo, neigh,
                                              wb_hi, wb_lo, w1_hi, w1_lo,
                                              gb_w, gb_b, gs_w, gs_b, out, N);
}
